// ContrastiveLoss_23021024707002
// MI455X (gfx1250) — matrix operand order repaired and checked by reading, not compiled, not run on hardware
//
#include <hip/hip_runtime.h>
#include <hip/hip_bf16.h>

// ContrastiveLoss on MI455X (gfx1250, wave32).
// HBM-bound: 256 MB streamed -> ~11us floor @ 23.3 TB/s. The 5 per-pair
// reductions (2 cross dots + 3 squared norms) for a 16-pair tile are the
// diagonals of five 16x16 Gram products, accumulated with
// v_wmma_f32_16x16x32_f16 over K=512. One wave per tile, 2048 waves total.

typedef __attribute__((ext_vector_type(16))) _Float16 v16h;
typedef __attribute__((ext_vector_type(8)))  float    v8f;
typedef __attribute__((ext_vector_type(4)))  float    v4f;

#define CL_D            512
#define CL_NPAIRS       32768
#define CL_TILE         16
#define WAVES_PER_BLOCK 8

// A-operand (16x32 f16, ISA 7.12.2): lanes 0-15 -> M=lane, K=k0+{0..7,16..23};
// lanes 16-31 -> M=lane-16, K=k0+{8..15,24..31}. Four aligned v4f loads + cvt.
__device__ inline v16h loadA_f16(const float* __restrict__ row, int k0, int lane) {
  const int ka = k0 + ((lane & 16) ? 8 : 0);
  v4f f0 = *reinterpret_cast<const v4f*>(row + ka);
  v4f f1 = *reinterpret_cast<const v4f*>(row + ka + 4);
  v4f f2 = *reinterpret_cast<const v4f*>(row + ka + 16);
  v4f f3 = *reinterpret_cast<const v4f*>(row + ka + 20);
  v16h a;
#pragma unroll
  for (int i = 0; i < 4; ++i) {
    a[i]      = (_Float16)f0[i];
    a[i + 4]  = (_Float16)f1[i];
    a[i + 8]  = (_Float16)f2[i];
    a[i + 12] = (_Float16)f3[i];
  }
  return a;
}

// Runtime extract from v8f accumulator without dynamic indexing (8-way cndmask).
__device__ inline float vsel8(v8f c, int ci) {
  float r = c[0];
#pragma unroll
  for (int i = 1; i < 8; ++i) r = (ci == i) ? c[i] : r;
  return r;
}

__global__ __launch_bounds__(WAVES_PER_BLOCK * 32)
void ContrastiveLoss_wmma_kernel(const float* __restrict__ q,
                                 const float* __restrict__ e,
                                 float* __restrict__ out) {
  const int lane = threadIdx.x & 31;
  const int wave = threadIdx.x >> 5;
  const int tile = blockIdx.x * WAVES_PER_BLOCK + wave;  // 0..2047
  const int p0   = tile * CL_TILE;                       // first pair of tile

  // A-operand row for this lane (M = lane&15):
  const int mA = lane & 15;
  const float* __restrict__ xrow  = q + (long)(2 * (p0 + mA)) * CL_D;  // x1 rows
  const float* __restrict__ y1row = e + (long)(2 * (p0 + mA)) * CL_D;  // y1 rows
  const float* __restrict__ y2row = y1row + CL_D;                      // y2 rows

  // Tile base pointers for the B-operand column gathers:
  const float* __restrict__ qb = q + (long)(2 * p0) * CL_D;
  const float* __restrict__ eb = e + (long)(2 * p0) * CL_D;

  v8f cXX = {}, cXY1 = {}, cXY2 = {}, cY1 = {}, cY2 = {};

  for (int k0 = 0; k0 < CL_D; k0 += 32) {
    if (k0 + 32 < CL_D) {  // uniform branch: EXEC stays all-ones for WMMA
      __builtin_prefetch(xrow  + k0 + 32, 0, 1);   // global_prefetch_b8
      __builtin_prefetch(y1row + k0 + 32, 0, 1);
      __builtin_prefetch(y2row + k0 + 32, 0, 1);
    }

    v16h aX  = loadA_f16(xrow,  k0, lane);
    v16h aY1 = loadA_f16(y1row, k0, lane);
    v16h aY2 = loadA_f16(y2row, k0, lane);

    // B-operand (32x16 f16) = M^T: lane holds column N=lane&15, i.e.
    // b[j] = M[lane&15][k], k in the same order as the A operand.
    const int c = k0 + ((lane & 16) ? 8 : 0);
    v16h bX, bY1, bY2;
#pragma unroll
    for (int j = 0; j < 16; ++j) {
      const long r = (long)(2 * (lane & 15)) * CL_D + j + ((j & 8) ? 8 : 0);
      bX[j]  = (_Float16)qb[r + c];            // x1 row lane&15
      bY1[j] = (_Float16)eb[r + c];            // y1 row lane&15
      bY2[j] = (_Float16)eb[r + c + CL_D];     // y2 row lane&15
    }

    // D = A*B + C ; (neg_a, A, neg_b, B, c_mod, C, reuse_a, reuse_b)
    cXX  = __builtin_amdgcn_wmma_f32_16x16x32_f16(false, aX,  false, bX,  (short)0, cXX,  false, false);
    cXY1 = __builtin_amdgcn_wmma_f32_16x16x32_f16(false, aX,  false, bY1, (short)0, cXY1, false, false);
    cXY2 = __builtin_amdgcn_wmma_f32_16x16x32_f16(false, aX,  false, bY2, (short)0, cXY2, false, false);
    cY1  = __builtin_amdgcn_wmma_f32_16x16x32_f16(false, aY1, false, bY1, (short)0, cY1,  false, false);
    cY2  = __builtin_amdgcn_wmma_f32_16x16x32_f16(false, aY2, false, bY2, (short)0, cY2,  false, false);
  }

  // Diagonal (m,m) of the 16x16 f32 C layout:
  //   m in 0..7  -> lane m,      VGPR m
  //   m in 8..15 -> lane m+16,   VGPR m-8
  const bool active = (lane < 8) || (lane >= 24);
  if (active) {
    const int ci = lane & 7;
    float xx   = vsel8(cXX,  ci);
    float xy1  = vsel8(cXY1, ci);
    float xy2  = vsel8(cXY2, ci);
    float y1y1 = vsel8(cY1,  ci);
    float y2y2 = vsel8(cY2,  ci);

    const float EPS = 1e-12f;
    float nx = fmaxf(sqrtf(xx),   EPS);
    float n1 = fmaxf(sqrtf(y1y1), EPS);
    float n2 = fmaxf(sqrtf(y2y2), EPS);
    float pos = xy1 / (nx * n1);
    float neg = xy2 / (nx * n2);
    // loss = -log_softmax([pos,neg])[0], numerically stable
    float mx  = fmaxf(pos, neg);
    float lse = mx + logf(expf(pos - mx) + expf(neg - mx));
    float loss = lse - pos;
    atomicAdd(out, loss * (1.0f / (float)CL_NPAIRS));
  }
}

__global__ void ContrastiveLoss_zero_kernel(float* out) {
  if (threadIdx.x == 0) out[0] = 0.0f;
}

extern "C" void kernel_launch(void* const* d_in, const int* in_sizes, int n_in,
                              void* d_out, int out_size, void* d_ws, size_t ws_size,
                              hipStream_t stream) {
  (void)in_sizes; (void)n_in; (void)out_size; (void)d_ws; (void)ws_size;
  const float* q = (const float*)d_in[0];   // query  [65536,512] fp32
  const float* e = (const float*)d_in[1];   // embed  [65536,512] fp32
  float* out = (float*)d_out;               // scalar fp32

  // d_out is poisoned; zero it every call (graph-replay deterministic).
  ContrastiveLoss_zero_kernel<<<1, 32, 0, stream>>>(out);

  const int tiles  = CL_NPAIRS / CL_TILE;        // 2048 tiles, 1 wave each
  const int blocks = tiles / WAVES_PER_BLOCK;    // 256 blocks x 8 waves
  ContrastiveLoss_wmma_kernel<<<blocks, WAVES_PER_BLOCK * 32, 0, stream>>>(q, e, out);
}
